// CAM_ModuleM_3289944949429
// MI455X (gfx1250) — matrix operand order repaired and checked by reading, not compiled, not run on hardware
//
#include <hip/hip_runtime.h>

// ---------------------------------------------------------------------------
// CAM module for MI455X (gfx1250, wave32, WMMA bf16 16x16x32)
//   energy = X^T X  (per batch)           -> bf16 WMMA GEMM, fp32 accum
//   attn   = softmax(rowmax - energy)     == exp(rowmin - e)/sum (stable form)
//   out    = gamma * (X @ attn) + x       -> bf16 WMMA GEMM, fused epilogue
//
// LDS tiles are stored PRE-PERMUTED into the WMMA per-lane register layout:
// a fragment load is one v16bf load (2x ds_load_b128), no repacking.
// Fill path: global_load_b128 (4x4 blocks), v_cvt_pk_bf16_f32 packing
// (via convertvector), v_perm_b32 row-interleave for bf16 B tiles, and
// ds_store_b64 commits. Next slab's global loads overlap current WMMAs.
// Workspace: [0,64MB) fp32 energy, [64MB,96MB) bf16 attention (L2-resident).
// ---------------------------------------------------------------------------

typedef __attribute__((ext_vector_type(16))) __bf16 v16bf;
typedef __attribute__((ext_vector_type(8)))  float  v8f;
typedef __attribute__((ext_vector_type(2)))  float  v2f;
typedef __attribute__((ext_vector_type(2)))  __bf16 v2bf;

#define BATCH 16
#define CDIM  512
#define NDIM  1024

#define TILE 128          // block output tile: TILE x TILE
#define KB   32           // K-slab depth (== WMMA K)
// packed LDS tile: 8 groups (of 16 rows/cols) x 32 lanes x 8 u32 = 2048 u32
#define PK_U32 (8 * 32 * 8)

// one v_cvt_pk_bf16_f32
__device__ __forceinline__ unsigned pack2bf(float lo, float hi) {
    v2f f = {lo, hi};
    v2bf b = __builtin_convertvector(f, v2bf);
    return __builtin_bit_cast(unsigned, b);
}

__device__ __forceinline__ float f4c(const float4& v, int c) {
    return reinterpret_cast<const float*>(&v)[c];
}

// A-matrix packed u32 index for element pair (i = M row-in-tile, k even)
// For k % 4 == 0: a_idx(i,k)+1 == a_idx(i,k+2)  (same lane, j+1) -> uint2 ok
__device__ __forceinline__ int a_idx(int i, int k) {
    const int g    = i >> 4;
    const int lane = (i & 15) + 16 * ((k >> 3) & 1);
    const int j    = 4 * ((k >> 4) & 1) + ((k >> 1) & 3);
    return (g * 32 + lane) * 8 + j;
}

// B-matrix packed u32 index for element pair (i = N col-in-tile, k even)
// For k % 4 == 0: b_idx(i,k)+1 == b_idx(i,k+2) -> uint2 ok
__device__ __forceinline__ int b_idx(int i, int k) {
    const int g    = i >> 4;
    const int lane = (i & 15) + 16 * ((k >> 3) & 1);
    const int j    = 4 * ((k >> 4) & 1) + ((k >> 1) & 3);
    return (g * 32 + lane) * 8 + j;
}

__device__ __forceinline__ v16bf frag_of(const unsigned* P, int g) {
    const int lane = threadIdx.x & 31;
    return reinterpret_cast<const v16bf*>(P)[g * 32 + lane];
}

// ---- Kernel 1: energy[b] = X^T X ------------------------------------------
// grid = (N/128, N/128, B), block = 256 (8 waves: 2 along M x 4 along N)
__global__ __launch_bounds__(256) void gram_gemm1(const float* __restrict__ x,
                                                  float* __restrict__ energy) {
    __shared__ __align__(32) unsigned Asp[PK_U32];
    __shared__ __align__(32) unsigned Bsp[PK_U32];

    const int b     = blockIdx.z;
    const int nBase = blockIdx.y * TILE;   // energy rows (n)
    const int mBase = blockIdx.x * TILE;   // energy cols (m)
    const float* xb = x + (size_t)b * CDIM * NDIM;

    const int tid  = threadIdx.x;
    const int lane = tid & 31;
    const int wave = tid >> 5;
    const int wm   = wave & 1;   // 0..1 -> 64 rows
    const int wn   = wave >> 1;  // 0..3 -> 32 cols

    // fill decomposition: each thread owns a 4(k) x 4(i) block of both slabs
    const int kq = tid >> 5;          // 0..7  -> k = 4*kq
    const int q  = tid & 31;          // 0..31 -> i0 = 4*q
    const int kk = 4 * kq;

    v8f acc[4][2];
#pragma unroll
    for (int i = 0; i < 4; ++i)
#pragma unroll
        for (int j = 0; j < 2; ++j) acc[i][j] = (v8f)0.0f;

    float4 ar[4], br[4];   // rows kk..kk+3, 4 cols each (global_load_b128)

#pragma unroll
    for (int r = 0; r < 4; ++r) {   // prologue: slab k0 = 0
        const size_t row = (size_t)(kk + r) * NDIM;
        ar[r] = reinterpret_cast<const float4*>(xb + row + nBase)[q];
        br[r] = reinterpret_cast<const float4*>(xb + row + mBase)[q];
    }

    for (int k0 = 0; k0 < CDIM; k0 += KB) {
        // commit staged 4x4 blocks into packed LDS (uint2 = ds_store_b64)
#pragma unroll
        for (int c = 0; c < 4; ++c) {
            const int i = 4 * q + c;
            uint2 av, bv;
            av.x = pack2bf(f4c(ar[0], c), f4c(ar[1], c));
            av.y = pack2bf(f4c(ar[2], c), f4c(ar[3], c));
            bv.x = pack2bf(f4c(br[0], c), f4c(br[1], c));
            bv.y = pack2bf(f4c(br[2], c), f4c(br[3], c));
            *reinterpret_cast<uint2*>(&Asp[a_idx(i, kk)]) = av;
            *reinterpret_cast<uint2*>(&Bsp[b_idx(i, kk)]) = bv;
        }
        __syncthreads();

        // issue global loads for next slab; they overlap the WMMAs below
        if (k0 + KB < CDIM) {
#pragma unroll
            for (int r = 0; r < 4; ++r) {
                const size_t row = (size_t)(k0 + KB + kk + r) * NDIM;
                ar[r] = reinterpret_cast<const float4*>(xb + row + nBase)[q];
                br[r] = reinterpret_cast<const float4*>(xb + row + mBase)[q];
            }
        }

#pragma unroll
        for (int tm = 0; tm < 4; ++tm) {
            const v16bf a = frag_of(Asp, wm * 4 + tm);
#pragma unroll
            for (int tn = 0; tn < 2; ++tn) {
                const v16bf bb = frag_of(Bsp, wn * 2 + tn);
                acc[tm][tn] = __builtin_amdgcn_wmma_f32_16x16x32_bf16(
                    false, a, false, bb, (short)0, acc[tm][tn], false, false);
            }
        }
        __syncthreads();
    }

    // D layout: VGPR v -> M = v + 8*half; lane&15 -> N
    float* eb = energy + (size_t)b * NDIM * NDIM;
    const int half = lane >> 4;
#pragma unroll
    for (int tm = 0; tm < 4; ++tm)
#pragma unroll
        for (int tn = 0; tn < 2; ++tn)
#pragma unroll
            for (int v = 0; v < 8; ++v) {
                const int r = nBase + wm * 64 + tm * 16 + v + 8 * half;
                const int c = mBase + wn * 32 + tn * 16 + (lane & 15);
                eb[(size_t)r * NDIM + c] = acc[tm][tn][v];
            }
}

// ---- Kernel 2: attn = softmax(rowmax - e) == exp(rowmin - e)/sum ----------
__global__ __launch_bounds__(256) void softmax_rows(const float* __restrict__ energy,
                                                    __bf16* __restrict__ attn) {
    const int row   = blockIdx.x;          // b*N + n
    const float* e  = energy + (size_t)row * NDIM;
    __bf16* a       = attn + (size_t)row * NDIM;
    const int tid   = threadIdx.x;

    float vals[4];
    float mn = 3.402823466e38f;
#pragma unroll
    for (int i = 0; i < 4; ++i) {
        vals[i] = e[tid + i * 256];
        mn = fminf(mn, vals[i]);
    }

    __shared__ float red[256];
    red[tid] = mn;
    __syncthreads();
    for (int s = 128; s > 0; s >>= 1) {
        if (tid < s) red[tid] = fminf(red[tid], red[tid + s]);
        __syncthreads();
    }
    mn = red[0];
    __syncthreads();

    float sum = 0.0f;
#pragma unroll
    for (int i = 0; i < 4; ++i) {
        vals[i] = __expf(mn - vals[i]);
        sum += vals[i];
    }
    red[tid] = sum;
    __syncthreads();
    for (int s = 128; s > 0; s >>= 1) {
        if (tid < s) red[tid] += red[tid + s];
        __syncthreads();
    }
    const float inv = 1.0f / red[0];
#pragma unroll
    for (int i = 0; i < 4; ++i)
        a[tid + i * 256] = (__bf16)(vals[i] * inv);
}

// ---- Kernel 3: out = gamma * (X @ attn) + x -------------------------------
// grid = (N/128, C/128, B), block = 256
__global__ __launch_bounds__(256) void out_gemm2(const float* __restrict__ x,
                                                 const __bf16* __restrict__ attn,
                                                 const float* __restrict__ gamma,
                                                 float* __restrict__ out) {
    __shared__ __align__(32) unsigned Asp[PK_U32];
    __shared__ __align__(32) unsigned Bsp[PK_U32];

    const int b     = blockIdx.z;
    const int cBase = blockIdx.y * TILE;   // out rows (c)
    const int mBase = blockIdx.x * TILE;   // out cols (m)
    const float*  xb = x    + (size_t)b * CDIM * NDIM;
    const __bf16* ab = attn + (size_t)b * NDIM * NDIM;

    const int tid  = threadIdx.x;
    const int lane = tid & 31;
    const int wave = tid >> 5;
    const int wm   = wave & 1;
    const int wn   = wave >> 1;

    // A fill: 4(i) x 4(k) block per thread; k fastest across threads
    const int iqA = tid >> 3;        // 0..31 -> i0 = 4*iqA
    const int kqA = tid & 7;         // 0..7  -> k  = 4*kqA
    const int i0A = 4 * iqA;
    const int kkA = 4 * kqA;

    v8f acc[4][2];
#pragma unroll
    for (int i = 0; i < 4; ++i)
#pragma unroll
        for (int j = 0; j < 2; ++j) acc[i][j] = (v8f)0.0f;

    float4 ra[4];            // A rows i0A..+3, k kkA..+3
    uint2  rbl[2], rbh[2];   // B rows k,k+1, 4 cols (bf16x4 = uint2)

#pragma unroll
    for (int r = 0; r < 4; ++r)   // prologue A, slab k0 = 0
        ra[r] = reinterpret_cast<const float4*>(
                    xb + (size_t)(cBase + i0A + r) * NDIM)[kqA];
#pragma unroll
    for (int it = 0; it < 2; ++it) {   // prologue B
        const int w  = tid + it * 256;
        const int k  = 2 * (w >> 5);   // 0..30
        const int jq = w & 31;
        const __bf16* rp = ab + (size_t)k * NDIM + mBase;
        rbl[it] = reinterpret_cast<const uint2*>(rp)[jq];
        rbh[it] = reinterpret_cast<const uint2*>(rp + NDIM)[jq];
    }

    for (int k0 = 0; k0 < NDIM; k0 += KB) {
        // commit A (cvt_pk + ds_store_b64)
#pragma unroll
        for (int r = 0; r < 4; ++r) {
            uint2 v;
            v.x = pack2bf(f4c(ra[r], 0), f4c(ra[r], 1));
            v.y = pack2bf(f4c(ra[r], 2), f4c(ra[r], 3));
            *reinterpret_cast<uint2*>(&Asp[a_idx(i0A + r, kkA)]) = v;
        }
        // commit B (v_perm row-interleave, 4x ds_store_b32)
#pragma unroll
        for (int it = 0; it < 2; ++it) {
            const int w  = tid + it * 256;
            const int k  = 2 * (w >> 5);
            const int j0 = 4 * (w & 31);
            const unsigned lo01 = __builtin_amdgcn_perm(rbh[it].x, rbl[it].x, 0x05040100u);
            const unsigned hi01 = __builtin_amdgcn_perm(rbh[it].x, rbl[it].x, 0x07060302u);
            const unsigned lo23 = __builtin_amdgcn_perm(rbh[it].y, rbl[it].y, 0x05040100u);
            const unsigned hi23 = __builtin_amdgcn_perm(rbh[it].y, rbl[it].y, 0x07060302u);
            Bsp[b_idx(j0 + 0, k)] = lo01;
            Bsp[b_idx(j0 + 1, k)] = hi01;
            Bsp[b_idx(j0 + 2, k)] = lo23;
            Bsp[b_idx(j0 + 3, k)] = hi23;
        }
        __syncthreads();

        if (k0 + KB < NDIM) {
#pragma unroll
            for (int r = 0; r < 4; ++r)
                ra[r] = reinterpret_cast<const float4*>(
                            xb + (size_t)(cBase + i0A + r) * NDIM + k0 + KB)[kqA];
#pragma unroll
            for (int it = 0; it < 2; ++it) {
                const int w  = tid + it * 256;
                const int k  = 2 * (w >> 5);
                const int jq = w & 31;
                const __bf16* rp = ab + (size_t)(k0 + KB + k) * NDIM + mBase;
                rbl[it] = reinterpret_cast<const uint2*>(rp)[jq];
                rbh[it] = reinterpret_cast<const uint2*>(rp + NDIM)[jq];
            }
        }

#pragma unroll
        for (int tm = 0; tm < 4; ++tm) {
            const v16bf a = frag_of(Asp, wm * 4 + tm);
#pragma unroll
            for (int tn = 0; tn < 2; ++tn) {
                const v16bf bb = frag_of(Bsp, wn * 2 + tn);
                acc[tm][tn] = __builtin_amdgcn_wmma_f32_16x16x32_bf16(
                    false, a, false, bb, (short)0, acc[tm][tn], false, false);
            }
        }
        __syncthreads();
    }

    const float g = gamma[0];
    const int half = lane >> 4;
#pragma unroll
    for (int tm = 0; tm < 4; ++tm)
#pragma unroll
        for (int tn = 0; tn < 2; ++tn)
#pragma unroll
            for (int v = 0; v < 8; ++v) {
                const int r = cBase + wm * 64 + tm * 16 + v + 8 * half;
                const int c = mBase + wn * 32 + tn * 16 + (lane & 15);
                const size_t idx = ((size_t)b * CDIM + r) * NDIM + c;
                out[idx] = g * acc[tm][tn][v] + xb[(size_t)r * NDIM + c];
            }
}

// ---------------------------------------------------------------------------
extern "C" void kernel_launch(void* const* d_in, const int* in_sizes, int n_in,
                              void* d_out, int out_size, void* d_ws, size_t ws_size,
                              hipStream_t stream) {
    (void)in_sizes; (void)n_in; (void)out_size; (void)ws_size;
    const float* x     = (const float*)d_in[0];
    const float* gamma = (const float*)d_in[1];
    float* out         = (float*)d_out;

    char* ws       = (char*)d_ws;
    float*  energy = (float*)ws;                                        // 64 MB
    __bf16* attn   = (__bf16*)(ws + (size_t)BATCH * NDIM * NDIM * 4);   // 32 MB

    dim3 g1(NDIM / TILE, NDIM / TILE, BATCH);   // 8 x 8 x 16
    gram_gemm1<<<g1, 256, 0, stream>>>(x, energy);

    softmax_rows<<<BATCH * NDIM, 256, 0, stream>>>(energy, attn);

    dim3 g2(NDIM / TILE, CDIM / TILE, BATCH);   // 8 x 4 x 16
    out_gemm2<<<g2, 256, 0, stream>>>(x, attn, gamma, out);
}
